// MambaBlock_56805237457493
// MI455X (gfx1250) — matrix operand order repaired and checked by reading, not compiled, not run on hardware
//
#include <hip/hip_runtime.h>
#include <hip/hip_bf16.h>

// ---------------------------------------------------------------------------
// Mamba block for gfx1250 (MI455X).
// GEMMs: v_wmma_f32_16x16x32_f16, tiles staged in LDS in *fragment order* so
// each lane reads its operand as 2x ds_load_b128 (no scalar u16 gathers).
// Weights are pre-transposed+converted to f16 (NxK) once per launch so the
// GEMM K-loop streams 16B-contiguous operand data.
// Shapes: B=2, L=2048, D_MODEL=1024, D_INNER=2048, D_STATE=16, DT_RANK=64.
// ---------------------------------------------------------------------------

typedef __attribute__((ext_vector_type(16))) _Float16 v16h;
typedef __attribute__((ext_vector_type(8)))  _Float16 v8h;
typedef __attribute__((ext_vector_type(8)))  float    v8f;
typedef __attribute__((ext_vector_type(4)))  float    v4f;

#define BM 64
#define BN 64
#define BKK 32

__device__ __forceinline__ float sigmoidf_(float v) { return 1.0f / (1.0f + expf(-v)); }
__device__ __forceinline__ float softplusf_(float v) { return v > 20.0f ? v : log1pf(expf(v)); }
__device__ __forceinline__ float geluf_(float v) { return 0.5f * v * (1.0f + erff(v * 0.70710678118654752f)); }

// ---------------------------------------------------------------------------
// Weight prep: W (KxN, f32, row-major) -> Wt (NxK, f16, row-major).
// One-shot, ~17MB total: negligible vs 23.3 TB/s; also halves weight reads
// inside the GEMM K-loops.
// ---------------------------------------------------------------------------
__global__ void __launch_bounds__(256)
transpose_f16_kernel(const float* __restrict__ in, _Float16* __restrict__ out,
                     int K, int N) {
  size_t i = (size_t)blockIdx.x * 256 + threadIdx.x;
  if (i >= (size_t)K * N) return;
  int n = (int)(i / K);
  int k = (int)(i - (size_t)n * K);
  out[i] = (_Float16)in[(size_t)k * N + n];
}

// ---------------------------------------------------------------------------
// WMMA GEMM: C[M,N] = epi(A[M,lda] * Wt^T (+ bias)), Wt is (N,K) f16.
// 128 threads = 4 waves; block computes 64x64; wave owns a 16-row strip
// (4 WMMA tiles along N). K stepped by 32.
//
// LDS holds tiles in FRAGMENT order:
//   AsF[t*512 + lane*16 + e] : t = M-subtile (0..3), 16-bit A 16x32 layout
//     lane = (m&15) + 16*grp, e: k = (e&7) + 8*grp + 16*(e>=8)
//   BsF[j*512 + lane*16 + e] : j = N-subtile (0..3), 16-bit B 32x16 layout
//     lane = (n&15) + 16*grp, e: k = (e&7) + 8*grp + 16*(e>=8)
// so operand fetch per lane = one 32B contiguous read (2x ds_load_b128).
//
// EPI: 0 none, 1 +bias, 2 +bias->softplus, 3 gelu.
// Requires M%64==0, K%32==0; N ragged OK (bounds-checked).
// ---------------------------------------------------------------------------
template <int EPI>
__global__ void __launch_bounds__(128)
gemm_wmma_f16(const float* __restrict__ A, const _Float16* __restrict__ Wt,
              const float* __restrict__ bias, float* __restrict__ C,
              int M, int N, int K, int lda) {
  __shared__ alignas(32) _Float16 AsF[4 * 512];
  __shared__ alignas(32) _Float16 BsF[4 * 512];

  const int tid  = threadIdx.x;
  const int wave = tid >> 5;          // 0..3
  const int lane = tid & 31;
  const int grp  = lane >> 4;
  const int l16  = lane & 15;

  const int bm = blockIdx.y * BM;
  const int bn = blockIdx.x * BN;

  // --- B staging coords: one 16-half run per thread ---
  const int  bj    = tid >> 5;            // N-subtile 0..3
  const int  bl    = tid & 31;            // dest lane
  const int  bgrp  = bl >> 4;             // K-group ((k>>3)&1)
  const int  bncol = bj * 16 + (bl & 15); // column within 64-wide tile
  const bool bvalid = (bn + bncol) < N;
  const _Float16* wrow = Wt + (size_t)(bn + (bvalid ? bncol : 0)) * K;

  v8f acc[4];
#pragma unroll
  for (int j = 0; j < 4; ++j) { v8f z = {}; acc[j] = z; }

  for (int k0 = 0; k0 < K; k0 += BKK) {
    // ---- Stage A: 256 runs of 8 halves; 2 runs per thread. ----
#pragma unroll
    for (int r = 0; r < 2; ++r) {
      int ri  = tid + r * 128;        // 0..255
      int t   = ri >> 6;              // M-subtile 0..3
      int rem = ri & 63;
      int sl  = rem >> 1;             // dest lane 0..31
      int he  = rem & 1;              // half of the 16-elem fragment
      int m   = t * 16 + (sl & 15);
      int sg  = sl >> 4;
      int kl  = 8 * sg + 16 * he;     // contiguous 8-k run start
      const float* src = A + (size_t)(bm + m) * lda + (k0 + kl);
      v4f f0 = *(const v4f*)(src);
      v4f f1 = *(const v4f*)(src + 4);
      v8h h;
#pragma unroll
      for (int e = 0; e < 4; ++e) {
        h[e]     = (_Float16)f0[e];
        h[4 + e] = (_Float16)f1[e];
      }
      *(v8h*)&AsF[t * 512 + sl * 16 + he * 8] = h;
      if (k0 + BKK < K) __builtin_prefetch(src + BKK, 0, 1);
    }

    // ---- Stage B: one 16-half contiguous run per thread. ----
    {
      const _Float16* src = wrow + k0 + bgrp * 8;
      v8h h0 = *(const v8h*)(src);
      v8h h1 = *(const v8h*)(src + 16);
      if (!bvalid) {
#pragma unroll
        for (int e = 0; e < 8; ++e) { h0[e] = (_Float16)0.0f; h1[e] = (_Float16)0.0f; }
      }
      *(v8h*)&BsF[bj * 512 + bl * 16]     = h0;
      *(v8h*)&BsF[bj * 512 + bl * 16 + 8] = h1;
      if (k0 + BKK < K) __builtin_prefetch(src + BKK, 0, 1);
    }
    __syncthreads();

    // ---- Compute: operands are contiguous 32B per lane. ----
    v16h afrag = *(const v16h*)&AsF[wave * 512 + lane * 16];
#pragma unroll
    for (int j = 0; j < 4; ++j) {
      v16h bfrag = *(const v16h*)&BsF[j * 512 + lane * 16];
      acc[j] = __builtin_amdgcn_wmma_f32_16x16x32_f16(
          false, afrag, false, bfrag, (short)0, acc[j], false, false);
    }
    __syncthreads();
  }

  // ---- Store: C/D 16x16 f32 layout -> VGPR r holds M = r + 8*grp. ----
#pragma unroll
  for (int j = 0; j < 4; ++j) {
    int col = bn + j * 16 + l16;
    if (col < N) {
#pragma unroll
      for (int r = 0; r < 8; ++r) {
        int row = bm + wave * 16 + r + 8 * grp;
        float v = acc[j][r];
        if (EPI == 1) v += bias[col];
        if (EPI == 2) v = softplusf_(v + bias[col]);
        if (EPI == 3) v = geluf_(v);
        C[(size_t)row * N + col] = v;
      }
    }
  }
}

// ---------------------------------------------------------------------------
// LayerNorm over last dim (D=1024). One block of 256 threads per row.
// ---------------------------------------------------------------------------
__global__ void __launch_bounds__(256)
ln_kernel(const float* __restrict__ x, const float* __restrict__ gamma,
          const float* __restrict__ beta, float* __restrict__ out) {
  const int D = 1024;
  const int row = blockIdx.x;
  const float* xr = x + (size_t)row * D;
  float* orow = out + (size_t)row * D;
  __shared__ float sred[8];
  const int tid = threadIdx.x, lane = tid & 31, wv = tid >> 5;

  float s = 0.0f;
  for (int i = tid; i < D; i += 256) s += xr[i];
#pragma unroll
  for (int off = 16; off > 0; off >>= 1) s += __shfl_xor(s, off, 32);
  if (lane == 0) sred[wv] = s;
  __syncthreads();
  float t = (lane < 8) ? sred[lane] : 0.0f;
#pragma unroll
  for (int off = 4; off > 0; off >>= 1) t += __shfl_xor(t, off, 32);
  const float mean = __shfl(t, 0, 32) * (1.0f / D);
  __syncthreads();

  float v = 0.0f;
  for (int i = tid; i < D; i += 256) { float d = xr[i] - mean; v += d * d; }
#pragma unroll
  for (int off = 16; off > 0; off >>= 1) v += __shfl_xor(v, off, 32);
  if (lane == 0) sred[wv] = v;
  __syncthreads();
  float tv = (lane < 8) ? sred[lane] : 0.0f;
#pragma unroll
  for (int off = 4; off > 0; off >>= 1) tv += __shfl_xor(tv, off, 32);
  const float rstd = rsqrtf(__shfl(tv, 0, 32) * (1.0f / D) + 1e-5f);

  for (int i = tid; i < D; i += 256)
    orow[i] = (xr[i] - mean) * rstd * gamma[i] + beta[i];
}

// ---------------------------------------------------------------------------
// Causal depthwise conv (taps=4) over the xi half of xz, + bias + SiLU.
// xz: (B*L, 4096), xi half = cols [0,2048). Output xi: (B*L, 2048).
// ---------------------------------------------------------------------------
__global__ void __launch_bounds__(256)
conv_silu_kernel(const float* __restrict__ xz, const float* __restrict__ w,
                 const float* __restrict__ bias, float* __restrict__ xi) {
  const int L_ = 2048, DI = 2048, XZW = 4096;
  size_t i = (size_t)blockIdx.x * 256 + threadIdx.x;
  if (i >= (size_t)2 * L_ * DI) return;
  int d = (int)(i & (DI - 1));
  size_t bl = i >> 11;
  int l = (int)(bl & (L_ - 1));
  size_t b = bl >> 11;

  float acc = bias[d];
#pragma unroll
  for (int j = 0; j < 4; ++j) {
    int ls = l - 3 + j;
    if (ls >= 0)
      acc += w[d * 4 + j] * xz[(b * L_ + ls) * XZW + d];
  }
  xi[i] = acc * sigmoidf_(acc);
}

// ---------------------------------------------------------------------------
// Selective scan. One lane per (b,d,n): 16 lanes cooperate on one channel,
// 2 channels per wave. Sequential over L with shfl_xor tree-sum for y.
// xdb: (B*L, 96) with Bm at col 64, C at col 80.
// ---------------------------------------------------------------------------
__global__ void __launch_bounds__(256)
scan_kernel(const float* __restrict__ u, const float* __restrict__ dt,
            const float* __restrict__ xdb, const float* __restrict__ A_log,
            const float* __restrict__ Dp, float* __restrict__ y) {
  const int L_ = 2048, DI = 2048, NS = 16, XDBW = 96;
  const int lane = threadIdx.x & 31;
  const int wv = threadIdx.x >> 5;
  const int n = lane & 15;
  const int pair = lane >> 4;
  const int idx = (blockIdx.x * 8 + wv) * 2 + pair;   // 0..4095 = b*2048+d
  const int b = idx >> 11;
  const int d = idx & (DI - 1);

  const float Ad = -expf(A_log[d * NS + n]);
  const float Dd = Dp[d];
  float h = 0.0f;
  const size_t base_bl = (size_t)b * L_;

  for (int l = 0; l < L_; ++l) {
    const size_t bl = base_bl + l;
    const float ut  = u[bl * DI + d];
    const float dtt = dt[bl * DI + d];
    const float Bn  = xdb[bl * XDBW + 64 + n];
    const float Cn  = xdb[bl * XDBW + 80 + n];
    const float dA  = expf(dtt * Ad);
    h = dA * h + (dtt * ut) * Bn;
    float part = h * Cn;
    part += __shfl_xor(part, 1, 32);
    part += __shfl_xor(part, 2, 32);
    part += __shfl_xor(part, 4, 32);
    part += __shfl_xor(part, 8, 32);
    if (n == 0) y[bl * DI + d] = part + ut * Dd;
  }
}

// y <- y * silu(z), z = xz[:, 2048 + d]
__global__ void __launch_bounds__(256)
gate_kernel(float* __restrict__ y, const float* __restrict__ xz) {
  size_t i = (size_t)blockIdx.x * 256 + threadIdx.x;
  if (i >= (size_t)8388608) return;
  int d = (int)(i & 2047);
  size_t bl = i >> 11;
  float z = xz[bl * 4096 + 2048 + d];
  y[i] = y[i] * (z * sigmoidf_(z));
}

// out = g[:, :1024] * sigmoid(g[:, 1024:]) + skip(x)
__global__ void __launch_bounds__(256)
glu_residual_kernel(const float* __restrict__ g, const float* __restrict__ x,
                    float* __restrict__ out) {
  size_t i = (size_t)blockIdx.x * 256 + threadIdx.x;
  if (i >= (size_t)4194304) return;
  int c = (int)(i & 1023);
  size_t bl = i >> 10;
  float g1 = g[bl * 2048 + c];
  float g2 = g[bl * 2048 + 1024 + c];
  out[i] = g1 * sigmoidf_(g2) + x[i];
}

// ---------------------------------------------------------------------------
extern "C" void kernel_launch(void* const* d_in, const int* in_sizes, int n_in,
                              void* d_out, int out_size, void* d_ws, size_t ws_size,
                              hipStream_t stream) {
  const float* x          = (const float*)d_in[0];
  const float* ln_gamma   = (const float*)d_in[1];
  const float* ln_beta    = (const float*)d_in[2];
  const float* in_proj_w  = (const float*)d_in[3];   // (1024, 4096)
  const float* conv_w     = (const float*)d_in[4];   // (2048, 4)
  const float* conv_b     = (const float*)d_in[5];
  const float* x_proj_w   = (const float*)d_in[6];   // (2048, 96)
  const float* dt_proj_w  = (const float*)d_in[7];   // (64, 2048)
  const float* dt_proj_b  = (const float*)d_in[8];
  const float* A_log      = (const float*)d_in[9];   // (2048, 16)
  const float* Dp         = (const float*)d_in[10];
  const float* out_proj_w = (const float*)d_in[11];  // (2048, 1024)
  const float* glu_w      = (const float*)d_in[12];  // (1024, 2048)
  const float* glu_b      = (const float*)d_in[13];

  const int M = 4096;  // B*L

  // Workspace layout (floats). Reuse: gate in-place on y; GELU out -> ws_ln;
  // GLU GEMM out -> ws_dt (dt dead after scan). f16 weight copies at the end.
  float* ws     = (float*)d_ws;
  float* ws_ln  = ws;                        // 4096*1024  =  4,194,304
  float* ws_xz  = ws_ln  + (size_t)4194304;  // 4096*4096  = 16,777,216
  float* ws_xi  = ws_xz  + (size_t)16777216; // 4096*2048  =  8,388,608
  float* ws_xdb = ws_xi  + (size_t)8388608;  // 4096*96    =    393,216
  float* ws_dt  = ws_xdb + (size_t)393216;   // 4096*2048  =  8,388,608
  float* ws_y   = ws_dt  + (size_t)8388608;  // 4096*2048  =  8,388,608

  _Float16* wt_in  = (_Float16*)(ws_y + (size_t)8388608); // (4096,1024) f16
  _Float16* wt_x   = wt_in  + (size_t)4194304;            // (96,2048)
  _Float16* wt_dt  = wt_x   + (size_t)196608;             // (2048,64)
  _Float16* wt_out = wt_dt  + (size_t)131072;             // (1024,2048)
  _Float16* wt_glu = wt_out + (size_t)2097152;            // (2048,1024)

  dim3 gblk(128);

  // 0) Weight prep (transpose + f16 convert)
  transpose_f16_kernel<<<(4194304 + 255) / 256, 256, 0, stream>>>(in_proj_w,  wt_in,  1024, 4096);
  transpose_f16_kernel<<<(196608  + 255) / 256, 256, 0, stream>>>(x_proj_w,   wt_x,   2048, 96);
  transpose_f16_kernel<<<(131072  + 255) / 256, 256, 0, stream>>>(dt_proj_w,  wt_dt,  64,   2048);
  transpose_f16_kernel<<<(2097152 + 255) / 256, 256, 0, stream>>>(out_proj_w, wt_out, 2048, 1024);
  transpose_f16_kernel<<<(2097152 + 255) / 256, 256, 0, stream>>>(glu_w,      wt_glu, 1024, 2048);

  // 1) LayerNorm
  ln_kernel<<<M, 256, 0, stream>>>(x, ln_gamma, ln_beta, ws_ln);

  // 2) in_proj: (4096,1024) x (1024,4096) -> xz
  gemm_wmma_f16<0><<<dim3(4096 / BN, M / BM), gblk, 0, stream>>>(
      ws_ln, wt_in, nullptr, ws_xz, M, 4096, 1024, 1024);

  // 3) causal depthwise conv + SiLU -> xi
  conv_silu_kernel<<<(8388608 + 255) / 256, 256, 0, stream>>>(
      ws_xz, conv_w, conv_b, ws_xi);

  // 4) x_proj: (4096,2048) x (2048,96) -> xdb (dt|B|C)
  gemm_wmma_f16<0><<<dim3(2, M / BM), gblk, 0, stream>>>(
      ws_xi, wt_x, nullptr, ws_xdb, M, 96, 2048, 2048);

  // 5) dt_proj + bias + softplus: (4096,64 of 96) x (64,2048) -> dt
  gemm_wmma_f16<2><<<dim3(2048 / BN, M / BM), gblk, 0, stream>>>(
      ws_xdb, wt_dt, dt_proj_b, ws_dt, M, 2048, 64, 96);

  // 6) selective scan -> y
  scan_kernel<<<256, 256, 0, stream>>>(ws_xi, ws_dt, ws_xdb, A_log, Dp, ws_y);

  // 7) y *= silu(z)
  gate_kernel<<<(8388608 + 255) / 256, 256, 0, stream>>>(ws_y, ws_xz);

  // 8) out_proj + GELU: (4096,2048) x (2048,1024) -> g1 (reuse ws_ln)
  gemm_wmma_f16<3><<<dim3(1024 / BN, M / BM), gblk, 0, stream>>>(
      ws_y, wt_out, nullptr, ws_ln, M, 1024, 2048, 2048);

  // 9) glu + bias: (4096,1024) x (1024,2048) -> g (reuse ws_dt)
  gemm_wmma_f16<1><<<dim3(2048 / BN, M / BM), gblk, 0, stream>>>(
      ws_ln, wt_glu, glu_b, ws_dt, M, 2048, 1024, 1024);

  // 10) GLU combine + residual
  glu_residual_kernel<<<(4194304 + 255) / 256, 256, 0, stream>>>(
      ws_dt, x, (float*)d_out);
}
